// FeedForwardNetwork_10273561772405
// MI455X (gfx1250) — compile-verified, matrix operand order checked by reading, not run on hardware
//
#include <hip/hip_runtime.h>
#include <math.h>

typedef __bf16 bf16_t;
typedef __attribute__((ext_vector_type(16))) __bf16 v16bf;
typedef __attribute__((ext_vector_type(8)))  __bf16 v8bf;
typedef __attribute__((ext_vector_type(8)))  float   v8f;

#define NPB     16      // nodes per block (one WMMA M tile)
#define THREADS 256     // 8 waves (wave32)
#define IN_DIM  1920

#define IN0   512
#define IN1   256
#define IN2   128
#define MID0S 1024      // silu scalars
#define NG    768       // gates
#define MID0  1792      // MID0S + NG
#define M1    512       // mid mul l=1
#define M2    256       // mid mul l=2

// LDS pitches in bf16 elems: K+8 -> row pitch = 2K+16 bytes == 16 (mod 256) -> conflict-free b128 frags
#define PX0 (IN0 + 8)
#define PX1 (IN1 + 8)
#define PX2 (IN2 + 8)
#define PM0 (MID0S + 8)
#define PM1 (M1 + 8)
#define PM2 (M2 + 8)

// LDS bf16-element offsets
#define OFF_X0 0
#define OFF_X1 (OFF_X0 + NPB * PX0)
#define OFF_X2 (OFF_X1 + 3 * NPB * PX1)
#define OFF_M0 (OFF_X2 + 5 * NPB * PX2)
#define OFF_M1 (OFF_M0 + NPB * PM0)
#define OFF_M2 (OFF_M1 + 3 * NPB * PM1)
#define BF16_TOTAL (OFF_M2 + 5 * NPB * PM2)
#define SMEM_BYTES (2 * BF16_TOTAL + NPB * NG * 4 + NPB * 4)

// ---------------- weight prep: fp32 [K][V] -> bf16 [V][K], scaled by 1/sqrt(K) -----------
__global__ void prep_w(const float* __restrict__ W, bf16_t* __restrict__ Wt,
                       int K, int V, float scale, int total) {
  int idx = blockIdx.x * blockDim.x + threadIdx.x;
  if (idx >= total) return;
  int u = idx / V;                 // coalesced read of W[u][v]
  int v = idx - u * V;
  Wt[(size_t)v * K + u] = (bf16_t)(W[idx] * scale);
}

// ---------------- 16x16 output tile over full K reduction ------------------------------
// A in LDS: [16 rows (node)][K] bf16, row pitch PITCH.  B in global ws: [v][K] bf16.
// 16-bit A/B fragment layout: lane (m = L&15, g = L>>4): regs 0-3 = K{g*8..g*8+7},
// regs 4-7 = K{16+g*8..}.  B mirrors A with n = L&15.
template<int K, int PITCH>
__device__ __forceinline__ v8f gemm16(const bf16_t* __restrict__ A,
                                      const bf16_t* __restrict__ Wt,
                                      int v0, int lane) {
  const int m = lane & 15;
  const int g = lane >> 4;
  const bf16_t* ap = A + m * PITCH + g * 8;
  const bf16_t* bp = Wt + (size_t)(v0 + m) * K + g * 8;
  v8f c = {};
#pragma unroll
  for (int k0 = 0; k0 < K; k0 += 32) {
    union { v16bf v; v8bf h[2]; } a, b;
    a.h[0] = *(const v8bf*)(ap + k0);
    a.h[1] = *(const v8bf*)(ap + k0 + 16);
    b.h[0] = *(const v8bf*)(bp + k0);
    b.h[1] = *(const v8bf*)(bp + k0 + 16);
    c = __builtin_amdgcn_wmma_f32_16x16x32_bf16(false, a.v, false, b.v,
                                                (short)0, c, false, false);
  }
  return c;
}

// ---------------- fused gated-equivariant FFN, 16 nodes per workgroup -------------------
__global__ void __launch_bounds__(THREADS)
ffn_kernel(const float* __restrict__ x, const float* __restrict__ attr,
           const float* __restrict__ b1, const float* __restrict__ b2,
           const bf16_t* __restrict__ Wt1_0, const bf16_t* __restrict__ Wt1_1,
           const bf16_t* __restrict__ Wt1_2, const bf16_t* __restrict__ Wt2_0,
           const bf16_t* __restrict__ Wt2_1, const bf16_t* __restrict__ Wt2_2,
           float* __restrict__ out) {
  extern __shared__ char smem[];
  bf16_t* sb     = (bf16_t*)smem;
  float*  sgates = (float*)(smem + 2 * BF16_TOTAL);
  float*  sattr  = sgates + NPB * NG;

  const int tid   = threadIdx.x;
  const int lane  = tid & 31;
  const int wave  = tid >> 5;
  const int nbase = blockIdx.x * NPB;

  // ---- stage 0: load x tile (coalesced fp32), de-interleave per (l,m) to bf16 LDS ----
  for (int i = tid; i < NPB * IN_DIM; i += THREADS) {
    int n   = i / IN_DIM;
    int col = i - n * IN_DIM;
    bf16_t bv = (bf16_t)x[(size_t)(nbase + n) * IN_DIM + col];
    if (col < IN0) {
      sb[OFF_X0 + n * PX0 + col] = bv;
    } else if (col < IN0 + 3 * IN1) {
      int t = col - IN0;
      int u = t / 3, mm = t - 3 * u;
      sb[OFF_X1 + mm * NPB * PX1 + n * PX1 + u] = bv;
    } else {
      int t = col - (IN0 + 3 * IN1);
      int u = t / 5, mm = t - 5 * u;
      sb[OFF_X2 + mm * NPB * PX2 + n * PX2 + u] = bv;
    }
  }
  if (tid < NPB) sattr[tid] = attr[nbase + tid];
  __syncthreads();

  // ---- stage 1: h0 = x0 @ W1_0 (1792 cols) -> silu scalars + sigmoid gates ----
  for (int t = wave; t < MID0 / 16; t += 8) {
    int v0 = t * 16;
    v8f c = gemm16<IN0, PX0>(sb + OFF_X0, Wt1_0, v0, lane);
    int vcol = v0 + (lane & 15);
    int nlo  = (lane >> 4) * 8;
    float bv = b1[vcol];
#pragma unroll
    for (int r = 0; r < 8; ++r) {
      int node  = nlo + r;
      float val = c[r] * sattr[node] + bv;
      if (vcol < MID0S) {
        sb[OFF_M0 + node * PM0 + vcol] = (bf16_t)(val / (1.0f + __expf(-val))); // silu
      } else {
        sgates[node * NG + (vcol - MID0S)] = 1.0f / (1.0f + __expf(-val));      // sigmoid
      }
    }
  }
  __syncthreads();

  // ---- stage 2: h1 (3 m-slices, 512 cols) and h2 (5 m-slices, 256 cols), gated ----
  for (int j = wave; j < 96 + 80; j += 8) {
    if (j < 96) {
      int mm = j >> 5, t = j & 31, v0 = t * 16;
      v8f c = gemm16<IN1, PX1>(sb + OFF_X1 + mm * NPB * PX1, Wt1_1, v0, lane);
      int vcol = v0 + (lane & 15);
      int nlo  = (lane >> 4) * 8;
#pragma unroll
      for (int r = 0; r < 8; ++r) {
        int node  = nlo + r;
        float val = c[r] * sattr[node] * sgates[node * NG + vcol];
        sb[OFF_M1 + mm * NPB * PM1 + node * PM1 + vcol] = (bf16_t)val;
      }
    } else {
      int jj = j - 96;
      int mm = jj >> 4, t = jj & 15, v0 = t * 16;
      v8f c = gemm16<IN2, PX2>(sb + OFF_X2 + mm * NPB * PX2, Wt1_2, v0, lane);
      int vcol = v0 + (lane & 15);
      int nlo  = (lane >> 4) * 8;
#pragma unroll
      for (int r = 0; r < 8; ++r) {
        int node  = nlo + r;
        float val = c[r] * sattr[node] * sgates[node * NG + M1 + vcol];
        sb[OFF_M2 + mm * NPB * PM2 + node * PM2 + vcol] = (bf16_t)val;
      }
    }
  }
  __syncthreads();

  // ---- stage 3: fctp_2 + bias(0e) + residual ----
  for (int j = wave; j < 32 + 48 + 40; j += 8) {
    if (j < 32) {                                  // l=0: K=1024 -> 512 cols
      int v0 = j * 16;
      v8f c = gemm16<MID0S, PM0>(sb + OFF_M0, Wt2_0, v0, lane);
      int vcol = v0 + (lane & 15);
      int nlo  = (lane >> 4) * 8;
      float bv = b2[vcol];
#pragma unroll
      for (int r = 0; r < 8; ++r) {
        int node  = nlo + r;
        float val = c[r] * sattr[node] + bv;
        size_t gid = (size_t)(nbase + node) * IN_DIM + vcol;
        out[gid] = val + x[gid];
      }
    } else if (j < 80) {                           // l=1: K=512 -> 256 cols, 3 m-slices
      int jj = j - 32;
      int mm = jj >> 4, t = jj & 15, v0 = t * 16;
      v8f c = gemm16<M1, PM1>(sb + OFF_M1 + mm * NPB * PM1, Wt2_1, v0, lane);
      int vcol = v0 + (lane & 15);
      int nlo  = (lane >> 4) * 8;
#pragma unroll
      for (int r = 0; r < 8; ++r) {
        int node  = nlo + r;
        float val = c[r] * sattr[node];
        size_t gid = (size_t)(nbase + node) * IN_DIM + IN0 + vcol * 3 + mm;
        out[gid] = val + x[gid];
      }
    } else {                                       // l=2: K=256 -> 128 cols, 5 m-slices
      int jj = j - 80;
      int mm = jj >> 3, t = jj & 7, v0 = t * 16;
      v8f c = gemm16<M2, PM2>(sb + OFF_M2 + mm * NPB * PM2, Wt2_2, v0, lane);
      int vcol = v0 + (lane & 15);
      int nlo  = (lane >> 4) * 8;
#pragma unroll
      for (int r = 0; r < 8; ++r) {
        int node  = nlo + r;
        float val = c[r] * sattr[node];
        size_t gid = (size_t)(nbase + node) * IN_DIM + IN0 + 3 * IN1 + vcol * 5 + mm;
        out[gid] = val + x[gid];
      }
    }
  }
}

extern "C" void kernel_launch(void* const* d_in, const int* in_sizes, int n_in,
                              void* d_out, int out_size, void* d_ws, size_t ws_size,
                              hipStream_t stream) {
  (void)in_sizes; (void)n_in; (void)out_size; (void)ws_size;
  const float* x    = (const float*)d_in[0];
  const float* attr = (const float*)d_in[1];
  const float* W1_0 = (const float*)d_in[2];
  const float* W1_1 = (const float*)d_in[3];
  const float* W1_2 = (const float*)d_in[4];
  const float* b1   = (const float*)d_in[5];
  const float* W2_0 = (const float*)d_in[6];
  const float* W2_1 = (const float*)d_in[7];
  const float* W2_2 = (const float*)d_in[8];
  const float* b2   = (const float*)d_in[9];
  float* out = (float*)d_out;

  // bf16 transposed+scaled weights in workspace (~3.4 MB)
  bf16_t* Wt1_0 = (bf16_t*)d_ws;
  bf16_t* Wt1_1 = Wt1_0 + (size_t)MID0 * IN0;
  bf16_t* Wt1_2 = Wt1_1 + (size_t)M1 * IN1;
  bf16_t* Wt2_0 = Wt1_2 + (size_t)M2 * IN2;
  bf16_t* Wt2_1 = Wt2_0 + (size_t)IN0 * MID0S;
  bf16_t* Wt2_2 = Wt2_1 + (size_t)IN1 * M1;

  auto prep = [&](const float* W, bf16_t* Wt, int K, int V) {
    int total = K * V;
    prep_w<<<(total + 255) / 256, 256, 0, stream>>>(W, Wt, K, V,
                                                    1.0f / sqrtf((float)K), total);
  };
  prep(W1_0, Wt1_0, IN0,   MID0);
  prep(W1_1, Wt1_1, IN1,   M1);
  prep(W1_2, Wt1_2, IN2,   M2);
  prep(W2_0, Wt2_0, MID0S, IN0);
  prep(W2_1, Wt2_1, M1,    IN1);
  prep(W2_2, Wt2_2, M2,    IN2);

  static_assert(SMEM_BYTES <= 320 * 1024, "LDS budget exceeded");
  hipFuncSetAttribute((const void*)ffn_kernel,
                      hipFuncAttributeMaxDynamicSharedMemorySize, SMEM_BYTES);

  const int nblocks = 32768 / NPB;   // 2048
  ffn_kernel<<<nblocks, THREADS, SMEM_BYTES, stream>>>(
      x, attr, b1, b2, Wt1_0, Wt1_1, Wt1_2, Wt2_0, Wt2_1, Wt2_2, out);
}
